// QANet_78804059947438
// MI455X (gfx1250) — compile-verified, matrix operand order checked by reading, not run on hardware
//
#include <hip/hip_runtime.h>
#include <math.h>

// ---------------------------------------------------------------------------
// QANet forward for MI455X (gfx1250, wave32, WMMA).
// All matrix contractions run through a generic tiled WMMA GEMM
// (v_wmma_f32_16x16x32_f16, fp32 accumulate). Each wave owns a 32x32 C tile:
// 2 A-fragments x 2 B-fragments -> 4 WMMAs per K-step. The K loop is split
// into a branch-free main loop (unpredicated global_load_b64 + cvt_pk) and a
// single guarded remainder; edge tiles take a fully branchless clamped path.
// ---------------------------------------------------------------------------

typedef __attribute__((ext_vector_type(16))) _Float16 v16h;
typedef __attribute__((ext_vector_type(2)))  _Float16 v2h;
typedef __attribute__((ext_vector_type(8)))  float    v8f;
typedef __attribute__((ext_vector_type(2)))  float    v2f;

#define B_SZ 16
#define D_M  128
#define L_C  400
#define L_Q  50
#define N_H  8
#define D_K  16
#define C_L  16
#define D_CH 64
#define D_WD 300
#define C_IN 428            // D_M + D_WD
#define A_SZ (B_SZ*L_C*D_M) // 819200 floats, one full context activation
#define NEGV (-1e30f)

// ---------------------------------------------------------------------------
// WMMA fragment K-index mapping (CDNA5 ISA 7.12.2, 16-bit A 16x32):
//  lanes 0-15 : elements 0..7 -> K 0..7,  elements 8..15 -> K 16..23
//  lanes 16-31: elements 0..7 -> K 8..15, elements 8..15 -> K 24..31
// B 32x16 mirrors A with N = lane%16.
// ---------------------------------------------------------------------------
__device__ __forceinline__ int wmma_kmap(int j, int lane) {
  return (j & 7) + ((j >> 3) << 4) + ((lane >> 4) << 3);
}

// branchless guarded load: clamp index, select zero. No EXEC manipulation.
__device__ __forceinline__ float ldg_guard(const float* __restrict__ p, long idx, bool ok) {
  float v = p[ok ? idx : 0];
  return ok ? v : 0.f;
}

// load 8+8 contiguous f32 (two runs, second at +16) into a v16h fragment,
// packing pairs with v_cvt_pk_rtz_f16_f32.
__device__ __forceinline__ void load_frag_contig(const float* __restrict__ base, v16h& f) {
  const v2f* p = (const v2f*)base;          // 8-byte aligned: all ldas are even
#pragma unroll
  for (int j = 0; j < 4; ++j) {
    v2f x0 = p[j];                          // K offsets 2j, 2j+1
    v2f x1 = p[j + 8];                      // K offsets 16+2j, 16+2j+1
    v2h h0 = __builtin_bit_cast(v2h, __builtin_amdgcn_cvt_pkrtz(x0.x, x0.y));
    v2h h1 = __builtin_bit_cast(v2h, __builtin_amdgcn_cvt_pkrtz(x1.x, x1.y));
    f[2 * j]     = h0.x;  f[2 * j + 1] = h0.y;
    f[2 * j + 8] = h1.x;  f[2 * j + 9] = h1.y;
  }
}

// Generic batched GEMM:  C = act(alpha * A x B (+bias))
// Per-operand batch offset: off(z) = (z/div)*outer + (z%div)*inner
// TB: B element (k,n) at B[n*ldb+k] (transposed) else B[k*ldb+n].
// ACT (runtime): 0 none, 1 relu, 2 sigmoid.  One wave => 32x32 C tile.
template<bool TB>
__global__ __launch_bounds__(128)
void k_gemm(const float* __restrict__ A, int lda, int adiv, long aouter, long ainner,
            const float* __restrict__ Bp, int ldb, int bdiv, long bouter, long binner,
            float* __restrict__ C, int ldc, int cdiv_, long couter, long cinner,
            const float* __restrict__ bias, int M, int N, int K, float alpha, int ACT)
{
  const int z = blockIdx.z;
  const float* Az = A  + (long)(z / adiv)  * aouter + (long)(z % adiv)  * ainner;
  const float* Bz = Bp + (long)(z / bdiv)  * bouter + (long)(z % bdiv)  * binner;
  float*       Cz = C  + (long)(z / cdiv_) * couter + (long)(z % cdiv_) * cinner;

  const int wave = threadIdx.x >> 5, lane = threadIdx.x & 31;
  const int tm = blockIdx.x, tn = blockIdx.y * 4 + wave;
  const int Nt = (N + 31) >> 5;
  if (tn >= Nt) return;                      // uniform per wave

  const int r     = lane & 15, hi = lane >> 4;
  const int kbase = hi * 8;                  // lane-half K offset
  const int row0  = tm * 32 + r, row1 = row0 + 16;
  const int col0  = tn * 32 + r, col1 = col0 + 16;
  const bool interior = (tm * 32 + 32 <= M) && (tn * 32 + 32 <= N);

  v8f acc00 = {}, acc01 = {}, acc10 = {}, acc11 = {};

  auto loadA2 = [&](int k0, v16h& f0, v16h& f1) {
    load_frag_contig(Az + (long)row0 * lda + k0 + kbase, f0);
    load_frag_contig(Az + (long)row1 * lda + k0 + kbase, f1);
  };
  auto loadB2 = [&](int k0, v16h& f0, v16h& f1) {
    if (TB) {
      load_frag_contig(Bz + (long)col0 * ldb + k0 + kbase, f0);
      load_frag_contig(Bz + (long)col1 * ldb + k0 + kbase, f1);
    } else {
      const float* bp = Bz + (long)(k0 + kbase) * ldb;
#pragma unroll
      for (int j = 0; j < 8; ++j) {
        const long r0 = (long)j * ldb, r1 = (long)(j + 16) * ldb;
        f0[j] = (_Float16)bp[r0 + col0]; f0[j + 8] = (_Float16)bp[r1 + col0];
        f1[j] = (_Float16)bp[r0 + col1]; f1[j + 8] = (_Float16)bp[r1 + col1];
      }
    }
  };
  auto loadA2g = [&](int k0, v16h& f0, v16h& f1) {
#pragma unroll
    for (int j = 0; j < 16; ++j) {
      const int k = k0 + wmma_kmap(j, lane);
      f0[j] = (_Float16)ldg_guard(Az, (long)row0 * lda + k, (row0 < M) && (k < K));
      f1[j] = (_Float16)ldg_guard(Az, (long)row1 * lda + k, (row1 < M) && (k < K));
    }
  };
  auto loadB2g = [&](int k0, v16h& f0, v16h& f1) {
#pragma unroll
    for (int j = 0; j < 16; ++j) {
      const int k = k0 + wmma_kmap(j, lane);
      const bool ok0 = (col0 < N) && (k < K), ok1 = (col1 < N) && (k < K);
      f0[j] = (_Float16)(TB ? ldg_guard(Bz, (long)col0 * ldb + k, ok0)
                            : ldg_guard(Bz, (long)k * ldb + col0, ok0));
      f1[j] = (_Float16)(TB ? ldg_guard(Bz, (long)col1 * ldb + k, ok1)
                            : ldg_guard(Bz, (long)k * ldb + col1, ok1));
    }
  };
  auto mma4 = [&](const v16h& a0, const v16h& a1, const v16h& b0, const v16h& b1) {
    acc00 = __builtin_amdgcn_wmma_f32_16x16x32_f16(false, a0, false, b0, (short)0, acc00, false, false);
    acc01 = __builtin_amdgcn_wmma_f32_16x16x32_f16(false, a0, false, b1, (short)0, acc01, false, false);
    acc10 = __builtin_amdgcn_wmma_f32_16x16x32_f16(false, a1, false, b0, (short)0, acc10, false, false);
    acc11 = __builtin_amdgcn_wmma_f32_16x16x32_f16(false, a1, false, b1, (short)0, acc11, false, false);
  };

  const int kfull = K & ~31;
  if (interior) {
    for (int k0 = 0; k0 < kfull; k0 += 32) {   // branch-free steady state
      v16h a0, a1, b0, b1;
      loadA2(k0, a0, a1);
      loadB2(k0, b0, b1);
      if (k0 + 32 < kfull)                     // gfx1250 global_prefetch_b8
        __builtin_prefetch(Az + (long)row0 * lda + k0 + 32, 0, 1);
      mma4(a0, a1, b0, b1);
    }
    if (kfull < K) {                           // single guarded remainder step
      v16h a0, a1, b0, b1;
      loadA2g(kfull, a0, a1);
      loadB2g(kfull, b0, b1);
      mma4(a0, a1, b0, b1);
    }
  } else {
    for (int k0 = 0; k0 < K; k0 += 32) {       // edge tiles: branchless clamps
      v16h a0, a1, b0, b1;
      loadA2g(k0, a0, a1);
      loadB2g(k0, b0, b1);
      mma4(a0, a1, b0, b1);
    }
  }

  auto fin = [&](float v, int oc) {
    v *= alpha;
    if (bias) v += bias[oc];
    if (ACT == 1) v = fmaxf(v, 0.f);
    else if (ACT == 2) v = 1.f / (1.f + expf(-v));
    return v;
  };
  if (interior) {
#pragma unroll
    for (int rr = 0; rr < 8; ++rr) {           // C layout: row = rr + 8*(lane/16)
      const int o = tm * 32 + hi * 8 + rr;
      Cz[(long)o * ldc + col0]        = fin(acc00[rr], col0);
      Cz[(long)o * ldc + col1]        = fin(acc01[rr], col1);
      Cz[(long)(o + 16) * ldc + col0] = fin(acc10[rr], col0);
      Cz[(long)(o + 16) * ldc + col1] = fin(acc11[rr], col1);
    }
  } else {
#pragma unroll
    for (int rr = 0; rr < 8; ++rr) {
      const int o = tm * 32 + hi * 8 + rr;
      if (o < M      && col0 < N) Cz[(long)o * ldc + col0]        = fin(acc00[rr], col0);
      if (o < M      && col1 < N) Cz[(long)o * ldc + col1]        = fin(acc01[rr], col1);
      if (o + 16 < M && col0 < N) Cz[(long)(o + 16) * ldc + col0] = fin(acc10[rr], col0);
      if (o + 16 < M && col1 < N) Cz[(long)(o + 16) * ldc + col1] = fin(acc11[rr], col1);
    }
  }
}

// ------------------------------- glue kernels ------------------------------

__global__ void k_mask(const int* __restrict__ wid, float* __restrict__ m, int n) {
  int i = blockIdx.x * blockDim.x + threadIdx.x;
  if (i < n) m[i] = (wid[i] != 0) ? 1.f : 0.f;
}

__global__ void k_add(float* __restrict__ d, const float* __restrict__ a,
                      const float* __restrict__ b, int n) {
  int i = blockIdx.x * blockDim.x + threadIdx.x;
  if (i < n) d[i] = a[i] + b[i];
}

// char depthwise 5x5 conv over (L, CL) incl. bias; out rows are GEMM-A rows.
__global__ void k_char_dw(const int* __restrict__ ids, const float* __restrict__ cm,
                          const float* __restrict__ dw, const float* __restrict__ dwb,
                          float* __restrict__ y, int L) {
  const long idx = (long)blockIdx.x * blockDim.x + threadIdx.x;
  const long total = (long)B_SZ * L * C_L * D_CH;
  if (idx >= total) return;
  const int c = (int)(idx & (D_CH - 1));
  long p = idx >> 6;
  const int t = (int)(p % C_L); p /= C_L;
  const int l = (int)(p % L);
  const int b = (int)(p / L);
  float s = dwb[c];
  for (int i = 0; i < 5; ++i) {
    const int ll = l + i - 2; if (ll < 0 || ll >= L) continue;
    for (int j = 0; j < 5; ++j) {
      const int tt = t + j - 2; if (tt < 0 || tt >= C_L) continue;
      const int id = ids[((long)b * L + ll) * C_L + tt];
      s += cm[(long)id * D_CH + c] * dw[c * 25 + i * 5 + j];
    }
  }
  y[idx] = s;
}

// relu(max over CL) into first D_M features of the 428-wide embedding buffer
__global__ void k_charmax(const float* __restrict__ z, float* __restrict__ emb, int L) {
  const long idx = (long)blockIdx.x * blockDim.x + threadIdx.x;
  const long total = (long)B_SZ * L * D_M;
  if (idx >= total) return;
  const int d = (int)(idx % D_M);
  const long bl = idx / D_M;
  float m = -3.4e38f;
  for (int t = 0; t < C_L; ++t)
    m = fmaxf(m, z[((long)bl * C_L + t) * D_M + d]);
  emb[bl * C_IN + d] = fmaxf(m, 0.f);
}

__global__ void k_wordgather(const int* __restrict__ wid, const float* __restrict__ wm,
                             float* __restrict__ emb, int L) {
  const long idx = (long)blockIdx.x * blockDim.x + threadIdx.x;
  const long total = (long)B_SZ * L * D_WD;
  if (idx >= total) return;
  const int j = (int)(idx % D_WD);
  const long bl = idx / D_WD;
  emb[bl * C_IN + D_M + j] = wm[(long)wid[bl] * D_WD + j];
}

// generic depthwise 1-D conv over length L on BLD ([b,l,c] c-contiguous)
__global__ void k_dw1d(const float* __restrict__ x, const float* __restrict__ dw,
                       const float* __restrict__ dwb, float* __restrict__ y,
                       int L, int C, int kw) {
  const long idx = (long)blockIdx.x * blockDim.x + threadIdx.x;
  const long total = (long)B_SZ * L * C;
  if (idx >= total) return;
  const int c = (int)(idx % C);
  const int l = (int)((idx / C) % L);
  const int b = (int)(idx / ((long)C * L));
  float s = dwb ? dwb[c] : 0.f;
  const int hw = kw >> 1;
  for (int i = 0; i < kw; ++i) {
    const int ll = l + i - hw;
    if (ll < 0 || ll >= L) continue;
    s += x[((long)b * L + ll) * C + c] * dw[c * kw + i];
  }
  y[idx] = s;
}

__global__ void k_hw(float* __restrict__ x, const float* __restrict__ g,
                     const float* __restrict__ nl, int n) {
  int i = blockIdx.x * blockDim.x + threadIdx.x;
  if (i < n) x[i] = g[i] * nl[i] + (1.f - g[i]) * x[i];
}

__global__ void k_posenc(const float* __restrict__ x, float* __restrict__ y, int L) {
  const long idx = (long)blockIdx.x * blockDim.x + threadIdx.x;
  const long total = (long)B_SZ * L * D_M;
  if (idx >= total) return;
  const int d = (int)(idx % D_M);
  const int l = (int)((idx / D_M) % L);
  float fr, ph;
  if (d & 1) { fr = -powf(10000.f, (1.f - (float)d) / (float)D_M); ph = 1.57079632679f; }
  else       { fr =  powf(10000.f, -(float)d / (float)D_M);        ph = 0.f; }
  y[idx] = x[idx] + sinf((float)l * fr + ph);
}

// layernorm over whole (D,L) per batch; gains/biases shaped (D,L)
__global__ void k_ln(const float* __restrict__ x, const float* __restrict__ g,
                     const float* __restrict__ b, float* __restrict__ y, int L) {
  __shared__ float ssum[256], ssq[256];
  const int bb = blockIdx.x;
  const float* xb = x + (long)bb * L * D_M;
  float*       yb = y + (long)bb * L * D_M;
  const int n = L * D_M;
  float s = 0.f, q = 0.f;
  for (int i = threadIdx.x; i < n; i += blockDim.x) { float v = xb[i]; s += v; q += v * v; }
  ssum[threadIdx.x] = s; ssq[threadIdx.x] = q;
  __syncthreads();
  for (int st = 128; st > 0; st >>= 1) {
    if (threadIdx.x < st) { ssum[threadIdx.x] += ssum[threadIdx.x + st];
                            ssq[threadIdx.x]  += ssq[threadIdx.x + st]; }
    __syncthreads();
  }
  const float mu  = ssum[0] / n;
  const float var = ssq[0] / n - mu * mu;
  const float inv = rsqrtf(var + 1e-5f);
  for (int i = threadIdx.x; i < n; i += blockDim.x) {
    const int l = i / D_M, d = i % D_M;
    yb[i] = (xb[i] - mu) * inv * g[(long)d * L + l] + b[(long)d * L + l];
  }
}

// attention softmax over the q axis (reference quirk: axis=2 of [b,h,q,m])
__global__ void k_smax_q(float* __restrict__ S, const float* __restrict__ mask, int L) {
  const int idx = blockIdx.x * blockDim.x + threadIdx.x;
  const int total = B_SZ * N_H * L;
  if (idx >= total) return;
  const int z = idx / L, m = idx % L, b = z / N_H;
  float* p = S + (long)z * L * L + m;
  const float* mk = mask + (long)b * L;
  float mx = -3.4e38f;
  for (int q = 0; q < L; ++q) { float v = p[(long)q * L] + (1.f - mk[q]) * NEGV; if (v > mx) mx = v; }
  float sum = 0.f;
  for (int q = 0; q < L; ++q) { float v = p[(long)q * L] + (1.f - mk[q]) * NEGV; sum += expf(v - mx); }
  const float inv = 1.f / sum;
  for (int q = 0; q < L; ++q) { float v = p[(long)q * L] + (1.f - mk[q]) * NEGV;
                                p[(long)q * L] = expf(v - mx) * inv; }
}

// CQ softmax over q (rows, contiguous) with qmask
__global__ void k_smax_row(const float* __restrict__ S, const float* __restrict__ qm,
                           float* __restrict__ O) {
  const int i = blockIdx.x * blockDim.x + threadIdx.x;
  if (i >= B_SZ * L_C) return;
  const int b = i / L_C;
  const float* p = S + (long)i * L_Q;
  const float* mk = qm + (long)b * L_Q;
  float* o = O + (long)i * L_Q;
  float mx = -3.4e38f;
  for (int q = 0; q < L_Q; ++q) { float v = p[q] + (1.f - mk[q]) * NEGV; if (v > mx) mx = v; }
  float sum = 0.f;
  for (int q = 0; q < L_Q; ++q) { float v = p[q] + (1.f - mk[q]) * NEGV; sum += expf(v - mx); }
  const float inv = 1.f / sum;
  for (int q = 0; q < L_Q; ++q) { float v = p[q] + (1.f - mk[q]) * NEGV; o[q] = expf(v - mx) * inv; }
}

// CQ softmax over c (columns, stride L_Q) with cmask
__global__ void k_smax_col(const float* __restrict__ S, const float* __restrict__ cm,
                           float* __restrict__ O) {
  const int i = blockIdx.x * blockDim.x + threadIdx.x;
  if (i >= B_SZ * L_Q) return;
  const int b = i / L_Q, q = i % L_Q;
  const float* p = S + (long)b * L_C * L_Q + q;
  const float* mk = cm + (long)b * L_C;
  float* o = O + (long)b * L_C * L_Q + q;
  float mx = -3.4e38f;
  for (int c = 0; c < L_C; ++c) { float v = p[(long)c * L_Q] + (1.f - mk[c]) * NEGV; if (v > mx) mx = v; }
  float sum = 0.f;
  for (int c = 0; c < L_C; ++c) { float v = p[(long)c * L_Q] + (1.f - mk[c]) * NEGV; sum += expf(v - mx); }
  const float inv = 1.f / sum;
  for (int c = 0; c < L_C; ++c) { float v = p[(long)c * L_Q] + (1.f - mk[c]) * NEGV;
                                  o[(long)c * L_Q] = expf(v - mx) * inv; }
}

__global__ void k_dotw(const float* __restrict__ x, const float* __restrict__ w,
                       float* __restrict__ o, int n) {
  const int i = blockIdx.x * blockDim.x + threadIdx.x;
  if (i >= n) return;
  const float* xp = x + (long)i * D_M;
  float s = 0.f;
  for (int d = 0; d < D_M; ++d) s += xp[d] * w[d];
  o[i] = s;
}

__global__ void k_scalew(const float* __restrict__ x, const float* __restrict__ w,
                         float* __restrict__ o, long n) {
  const long i = (long)blockIdx.x * blockDim.x + threadIdx.x;
  if (i < n) o[i] = x[i] * w[i % D_M];
}

__global__ void k_addS(float* __restrict__ S, const float* __restrict__ Cwc,
                       const float* __restrict__ Qwq) {
  const long i = (long)blockIdx.x * blockDim.x + threadIdx.x;
  const long total = (long)B_SZ * L_C * L_Q;
  if (i >= total) return;
  const int q = (int)(i % L_Q);
  const int c = (int)((i / L_Q) % L_C);
  const int b = (int)(i / ((long)L_Q * L_C));
  S[i] += Cwc[b * L_C + c] + Qwq[b * L_Q + q];
}

__global__ void k_concatX(const float* __restrict__ Ce, const float* __restrict__ Ab,
                          const float* __restrict__ Bb, float* __restrict__ X) {
  const long i = (long)blockIdx.x * blockDim.x + threadIdx.x;
  const long total = (long)B_SZ * L_C * 512;
  if (i >= total) return;
  const int s4 = (int)(i % 512);
  const long bl = i / 512;
  const int seg = s4 >> 7, d = s4 & 127;
  const float c = Ce[bl * D_M + d];
  float o;
  if      (seg == 0) o = c;
  else if (seg == 1) o = Ab[bl * D_M + d];
  else if (seg == 2) o = c * Ab[bl * D_M + d];
  else               o = c * Bb[bl * D_M + d];
  X[i] = o;
}

// span pointer: y = w . concat(Ma,Mb); masked log_softmax over L
__global__ void k_pointer(const float* __restrict__ Ma, const float* __restrict__ Mb,
                          const float* __restrict__ w, const float* __restrict__ mask,
                          float* __restrict__ out, int L) {
  __shared__ float red[512];
  const int b = blockIdx.x, l = threadIdx.x;
  float val = -3.4e38f;
  if (l < L) {
    const float* pa = Ma + ((long)b * L + l) * D_M;
    const float* pb = Mb + ((long)b * L + l) * D_M;
    float s = 0.f;
    for (int d = 0; d < D_M; ++d) s += w[d] * pa[d] + w[D_M + d] * pb[d];
    val = s + (1.f - mask[b * L + l]) * NEGV;
  }
  red[l] = (l < L) ? val : -3.4e38f;
  __syncthreads();
  for (int st = 256; st > 0; st >>= 1) {
    if (l < st) red[l] = fmaxf(red[l], red[l + st]);
    __syncthreads();
  }
  const float mx = red[0];
  __syncthreads();
  red[l] = (l < L) ? expf(val - mx) : 0.f;
  __syncthreads();
  for (int st = 256; st > 0; st >>= 1) {
    if (l < st) red[l] += red[l + st];
    __syncthreads();
  }
  const float sm = red[0];
  if (l < L) out[(long)b * L + l] = val - mx - logf(sm);
}

// ------------------------------- host side ---------------------------------

static inline int cdiv(int a, int b) { return (a + b - 1) / b; }

template<bool TB>
static void gemm1(const float* A, int lda, const float* Bp, int ldb,
                  float* C, int ldc, const float* bias,
                  int M, int N, int K, float alpha, int ACT, hipStream_t st) {
  dim3 g(cdiv(M, 32), cdiv(cdiv(N, 32), 4), 1);
  k_gemm<TB><<<g, 128, 0, st>>>(A, lda, 1, 0, 0, Bp, ldb, 1, 0, 0,
                                C, ldc, 1, 0, 0, bias, M, N, K, alpha, ACT);
}

struct EncP {
  const float *Wk, *Wo, *Wq, *Wv, *conv_dw, *conv_pw, *fc_b, *fc_w,
              *ln_b_b, *ln_b_g, *ln_e_b, *ln_e_g, *ln_s_b, *ln_s_g;
};

static void run_enc_block(const float* xin, float* xout, int L, const float* mask,
                          const EncP& P, int convnum, int kw,
                          float* r, float* t, float* u, float* v,
                          float* Qb, float* Kb, float* Vb, float* Sb, float* Hb,
                          hipStream_t st) {
  const int n = B_SZ * L * D_M;
  const int EB = cdiv(n, 256);
  k_posenc<<<EB, 256, 0, st>>>(xin, r, L);
  k_ln<<<B_SZ, 256, 0, st>>>(r, P.ln_b_g, P.ln_b_b, t, L);
  float* res = r; float* cur = t; float* sa = u; float* sb = v;
  for (int i = 0; i < convnum; ++i) {
    k_dw1d<<<EB, 256, 0, st>>>(cur, P.conv_dw + (long)i * D_M * kw, nullptr, sa, L, D_M, kw);
    gemm1<true>(sa, D_M, P.conv_pw + (long)i * D_M * D_M, D_M, sb, D_M,
                nullptr, B_SZ * L, D_M, D_M, 1.f, 1, st);
    k_add<<<EB, 256, 0, st>>>(sb, sb, res, n);
    k_ln<<<B_SZ, 256, 0, st>>>(sb, P.ln_s_g + (long)i * D_M * L,
                               P.ln_s_b + (long)i * D_M * L, cur, L);
    float* old = res; res = sb; sb = old;
  }
  const int Z = B_SZ * N_H;
  {
    dim3 g(cdiv(L, 32), 1, Z); // N=16 -> one tile column
    k_gemm<false><<<g, 128, 0, st>>>(cur, D_M, N_H, (long)L * D_M, 0,
        P.Wq, D_K, N_H, 0, (long)D_M * D_K, Qb, D_K, 1, (long)L * D_K, 0,
        nullptr, L, D_K, D_M, 1.f, 0);
    k_gemm<false><<<g, 128, 0, st>>>(cur, D_M, N_H, (long)L * D_M, 0,
        P.Wk, D_K, N_H, 0, (long)D_M * D_K, Kb, D_K, 1, (long)L * D_K, 0,
        nullptr, L, D_K, D_M, 1.f, 0);
    k_gemm<false><<<g, 128, 0, st>>>(cur, D_M, N_H, (long)L * D_M, 0,
        P.Wv, D_K, N_H, 0, (long)D_M * D_K, Vb, D_K, 1, (long)L * D_K, 0,
        nullptr, L, D_K, D_M, 1.f, 0);
  }
  {
    dim3 g(cdiv(L, 32), cdiv(cdiv(L, 32), 4), Z);
    k_gemm<true><<<g, 128, 0, st>>>(Qb, D_K, 1, (long)L * D_K, 0,
        Kb, D_K, 1, (long)L * D_K, 0, Sb, L, 1, (long)L * L, 0,
        nullptr, L, L, D_K, 0.25f, 0); // 1/sqrt(16)
  }
  k_smax_q<<<cdiv(Z * L, 256), 256, 0, st>>>(Sb, mask, L);
  {
    dim3 g(cdiv(L, 32), 1, Z);
    // head concat: C offset = (z/NH)*L*D + (z%NH)*16, ldc = D
    k_gemm<false><<<g, 128, 0, st>>>(Sb, L, 1, (long)L * L, 0,
        Vb, D_K, 1, (long)L * D_K, 0, Hb, D_M, N_H, (long)L * D_M, (long)D_K,
        nullptr, L, D_K, L, 1.f, 0);
  }
  gemm1<false>(Hb, D_M, P.Wo, D_M, sa, D_M, nullptr, B_SZ * L, D_M, D_M, 1.f, 0, st);
  k_add<<<EB, 256, 0, st>>>(sa, sa, res, n);
  k_ln<<<B_SZ, 256, 0, st>>>(sa, P.ln_e_g, P.ln_e_b, cur, L);
  gemm1<true>(cur, D_M, P.fc_w, D_M, sb, D_M, P.fc_b, B_SZ * L, D_M, D_M, 1.f, 1, st);
  k_add<<<EB, 256, 0, st>>>(xout, sb, sa, n);
}

extern "C" void kernel_launch(void* const* d_in, const int* in_sizes, int n_in,
                              void* d_out, int out_size, void* d_ws, size_t ws_size,
                              hipStream_t stream) {
  (void)in_sizes; (void)n_in; (void)out_size; (void)ws_size;
  // Inputs: Cwid, Ccid, Qwid, Qcid, then params flattened (alphabetical, recursive)
  const int* Cwid = (const int*)d_in[0];
  const int* Ccid = (const int*)d_in[1];
  const int* Qwid = (const int*)d_in[2];
  const int* Qcid = (const int*)d_in[3];
  auto F = [&](int i) { return (const float*)d_in[i]; };
  auto encAt = [&](int b) {
    EncP p = { F(b+0), F(b+1), F(b+2), F(b+3), F(b+4), F(b+5), F(b+6),
               F(b+7), F(b+8), F(b+9), F(b+10), F(b+11), F(b+12), F(b+13) };
    return p;
  };
  EncP cenc = encAt(4);               // c_enc.* (Wk,Wo,Wq,Wv,conv_dw,conv_pw,fc_b,fc_w,ln_*)
  const float* char_mat = F(18);
  const float* cq_w = F(19);
  const float *c1_dw = F(20), *c1_dwb = F(21), *c1_pw = F(22), *c1_pwb = F(23);
  const float *c2_dw = F(24), *c2_dwb = F(25), *c2_pw = F(26), *c2_pwb = F(27);
  const float *hw_gate_b = F(28), *hw_gate_w = F(29), *hw_lin_b = F(30), *hw_lin_w = F(31);
  EncP menc = encAt(32);
  EncP qenc = encAt(46);
  const float *res_dw = F(60), *res_pw = F(61), *w1 = F(62), *w2 = F(63), *word_mat = F(64);

  // --------------------------- workspace carve -----------------------------
  float* wsf = (float*)d_ws;
  size_t off = 0;
  auto alloc = [&](size_t n) { float* p = wsf + off; off += (n + 63) & ~(size_t)63; return p; };
  float* cmask = alloc(B_SZ * L_C);
  float* qmask = alloc(B_SZ * L_Q);
  float* embB  = alloc((size_t)B_SZ * L_C * C_IN);
  float* dwbuf = alloc((size_t)B_SZ * L_C * 512);
  float* charY = alloc((size_t)B_SZ * L_C * C_L * D_CH);
  float* charZ = alloc((size_t)B_SZ * L_C * C_L * D_M);
  float* xC    = alloc(A_SZ);
  float* xQ    = alloc(A_SZ);
  float* gbuf  = alloc(A_SZ);
  float* nlbuf = alloc(A_SZ);
  float* r  = alloc(A_SZ); float* t  = alloc(A_SZ);
  float* u  = alloc(A_SZ); float* v  = alloc(A_SZ);
  float* Qb = alloc(A_SZ); float* Kb = alloc(A_SZ); float* Vb = alloc(A_SZ);
  float* Sb = alloc((size_t)B_SZ * N_H * L_C * L_C);
  float* Hb = alloc(A_SZ);
  float* Ce = alloc(A_SZ); float* Qe = alloc(A_SZ);
  float* Cwc = alloc(B_SZ * L_C); float* Qwq = alloc(B_SZ * L_Q);
  float* Cm  = alloc(A_SZ);
  float* Scq = alloc((size_t)B_SZ * L_C * L_Q);
  float* S1  = alloc((size_t)B_SZ * L_C * L_Q);
  float* S2  = alloc((size_t)B_SZ * L_C * L_Q);
  float* Tcc = alloc((size_t)B_SZ * L_C * L_C);
  float* Abuf = alloc(A_SZ); float* Bbuf = alloc(A_SZ);
  float* Xc  = alloc((size_t)B_SZ * L_C * 512);
  float* M1w = alloc(A_SZ); float* Mtmp = alloc(A_SZ);
  float* M1s = alloc(A_SZ); float* M2s = alloc(A_SZ);

  // ------------------------------ masks ------------------------------------
  k_mask<<<cdiv(B_SZ * L_C, 256), 256, 0, stream>>>(Cwid, cmask, B_SZ * L_C);
  k_mask<<<cdiv(B_SZ * L_Q, 256), 256, 0, stream>>>(Qwid, qmask, B_SZ * L_Q);

  // ---------------------------- embedding ----------------------------------
  auto run_embedding = [&](const int* wid, const int* cid, int L, float* xout) {
    const long nchar = (long)B_SZ * L * C_L * D_CH;
    k_char_dw<<<(int)((nchar + 255) / 256), 256, 0, stream>>>(cid, char_mat, c2_dw, c2_dwb, charY, L);
    gemm1<true>(charY, D_CH, c2_pw, D_CH, charZ, D_M, c2_pwb,
                B_SZ * L * C_L, D_M, D_CH, 1.f, 0, stream);        // 64 -> 128 pointwise
    const int nmax = B_SZ * L * D_M;
    k_charmax<<<cdiv(nmax, 256), 256, 0, stream>>>(charZ, embB, L);
    const long nw = (long)B_SZ * L * D_WD;
    k_wordgather<<<(int)((nw + 255) / 256), 256, 0, stream>>>(wid, word_mat, embB, L);
    const long ndw = (long)B_SZ * L * C_IN;
    k_dw1d<<<(int)((ndw + 255) / 256), 256, 0, stream>>>(embB, c1_dw, c1_dwb, dwbuf, L, C_IN, 5);
    gemm1<true>(dwbuf, C_IN, c1_pw, C_IN, xout, D_M, c1_pwb,
                B_SZ * L, D_M, C_IN, 1.f, 0, stream);              // 428 -> 128 pointwise
    for (int i = 0; i < 2; ++i) {                                  // highway x2
      gemm1<true>(xout, D_M, hw_gate_w + (long)i * D_M * D_M, D_M, gbuf, D_M,
                  hw_gate_b + (long)i * D_M, B_SZ * L, D_M, D_M, 1.f, 2, stream);
      gemm1<true>(xout, D_M, hw_lin_w + (long)i * D_M * D_M, D_M, nlbuf, D_M,
                  hw_lin_b + (long)i * D_M, B_SZ * L, D_M, D_M, 1.f, 1, stream);
      k_hw<<<cdiv(nmax, 256), 256, 0, stream>>>(xout, gbuf, nlbuf, nmax);
    }
  };
  run_embedding(Cwid, Ccid, L_C, xC);
  run_embedding(Qwid, Qcid, L_Q, xQ);

  // ------------------------- embedding encoders ----------------------------
  run_enc_block(xC, Ce, L_C, cmask, cenc, 4, 7, r, t, u, v, Qb, Kb, Vb, Sb, Hb, stream);
  run_enc_block(xQ, Qe, L_Q, qmask, qenc, 4, 7, r, t, u, v, Qb, Kb, Vb, Sb, Hb, stream);

  // ---------------------------- CQ attention -------------------------------
  k_dotw<<<cdiv(B_SZ * L_C, 256), 256, 0, stream>>>(Ce, cq_w + D_M, Cwc, B_SZ * L_C);   // w_c
  k_dotw<<<cdiv(B_SZ * L_Q, 256), 256, 0, stream>>>(Qe, cq_w, Qwq, B_SZ * L_Q);         // w_q
  k_scalew<<<cdiv(B_SZ * L_C * D_M, 256), 256, 0, stream>>>(Ce, cq_w + 2 * D_M, Cm,
                                                            (long)B_SZ * L_C * D_M);    // C * w_m
  {
    dim3 g(cdiv(L_C, 32), 1, B_SZ);
    k_gemm<true><<<g, 128, 0, stream>>>(Cm, D_M, 1, (long)L_C * D_M, 0,
        Qe, D_M, 1, (long)L_Q * D_M, 0, Scq, L_Q, 1, (long)L_C * L_Q, 0,
        nullptr, L_C, L_Q, D_M, 1.f, 0);
  }
  k_addS<<<cdiv(B_SZ * L_C * L_Q, 256), 256, 0, stream>>>(Scq, Cwc, Qwq);
  k_smax_row<<<cdiv(B_SZ * L_C, 256), 256, 0, stream>>>(Scq, qmask, S1);
  k_smax_col<<<cdiv(B_SZ * L_Q, 256), 256, 0, stream>>>(Scq, cmask, S2);
  {
    dim3 g(cdiv(L_C, 32), 1, B_SZ); // A = S1 @ Q
    k_gemm<false><<<g, 128, 0, stream>>>(S1, L_Q, 1, (long)L_C * L_Q, 0,
        Qe, D_M, 1, (long)L_Q * D_M, 0, Abuf, D_M, 1, (long)L_C * D_M, 0,
        nullptr, L_C, D_M, L_Q, 1.f, 0);
  }
  {
    dim3 g(cdiv(L_C, 32), cdiv(cdiv(L_C, 32), 4), B_SZ); // T = S1 @ S2^T
    k_gemm<true><<<g, 128, 0, stream>>>(S1, L_Q, 1, (long)L_C * L_Q, 0,
        S2, L_Q, 1, (long)L_C * L_Q, 0, Tcc, L_C, 1, (long)L_C * L_C, 0,
        nullptr, L_C, L_C, L_Q, 1.f, 0);
  }
  {
    dim3 g(cdiv(L_C, 32), 1, B_SZ); // Bm = T @ C
    k_gemm<false><<<g, 128, 0, stream>>>(Tcc, L_C, 1, (long)L_C * L_C, 0,
        Ce, D_M, 1, (long)L_C * D_M, 0, Bbuf, D_M, 1, (long)L_C * D_M, 0,
        nullptr, L_C, D_M, L_C, 1.f, 0);
  }
  k_concatX<<<cdiv(B_SZ * L_C * 512, 256), 256, 0, stream>>>(Ce, Abuf, Bbuf, Xc);

  // ------------------------ resize conv (512 -> 128) -----------------------
  k_dw1d<<<cdiv(B_SZ * L_C * 512, 256), 256, 0, stream>>>(Xc, res_dw, nullptr, dwbuf, L_C, 512, 5);
  gemm1<true>(dwbuf, 512, res_pw, 512, M1w, D_M, nullptr, B_SZ * L_C, D_M, 512, 1.f, 0, stream);

  // ------------------------- 21 model encoder blocks -----------------------
  float* curM = M1w; float* tmpM = Mtmp;
  for (int i = 0; i < 7; ++i) {
    run_enc_block(curM, tmpM, L_C, cmask, menc, 2, 5, r, t, u, v, Qb, Kb, Vb, Sb, Hb, stream);
    float* sw = curM; curM = tmpM; tmpM = sw;
  }
  (void)hipMemcpyAsync(M1s, curM, sizeof(float) * A_SZ, hipMemcpyDeviceToDevice, stream);
  for (int i = 0; i < 7; ++i) {
    run_enc_block(curM, tmpM, L_C, cmask, menc, 2, 5, r, t, u, v, Qb, Kb, Vb, Sb, Hb, stream);
    float* sw = curM; curM = tmpM; tmpM = sw;
  }
  (void)hipMemcpyAsync(M2s, curM, sizeof(float) * A_SZ, hipMemcpyDeviceToDevice, stream);
  for (int i = 0; i < 7; ++i) {
    run_enc_block(curM, tmpM, L_C, cmask, menc, 2, 5, r, t, u, v, Qb, Kb, Vb, Sb, Hb, stream);
    float* sw = curM; curM = tmpM; tmpM = sw;
  }
  // curM now holds M3

  // --------------------------- span pointers -------------------------------
  float* out = (float*)d_out;
  k_pointer<<<B_SZ, 512, 0, stream>>>(M1s, M2s, w1, cmask, out, L_C);
  k_pointer<<<B_SZ, 512, 0, stream>>>(M1s, curM, w2, cmask, out + B_SZ * L_C, L_C);
}
